// TiledAttention_81166291959865
// MI455X (gfx1250) — matrix operand order repaired and checked by reading, not compiled, not run on hardware
//
#include <hip/hip_runtime.h>
#include <hip/hip_bf16.h>

typedef __attribute__((ext_vector_type(16))) _Float16 v16h;
typedef __attribute__((ext_vector_type(8)))  _Float16 v8h;
typedef __attribute__((ext_vector_type(4)))  _Float16 v4h;
typedef __attribute__((ext_vector_type(2)))  _Float16 v2h;
typedef __attribute__((ext_vector_type(8)))  float    v8f;
typedef __attribute__((ext_vector_type(4)))  float    f32x4;

#define NQ    4096   // sequence length
#define HD    128    // head dim
#define NB    4      // batch
#define BM    64     // query rows per workgroup (16 per wave)
#define BN    64     // keys per inner iteration (4 S-tiles)
#define NWAVE 4

#define KPAD (HD + 8)   // Ksh row stride (halves): 272B rows
#define VPAD (BN + 8)   // Vsh/Psh row stride (halves): 144B rows

#define DPP8_SEL(a,b,c,d,e,f,g,h) \
  ((a)|((b)<<3)|((c)<<6)|((d)<<9)|((e)<<12)|((f)<<15)|((g)<<18)|((h)<<21))

static __device__ __forceinline__ v16h cat8(v8h lo, v8h hi) {
  return __builtin_shufflevector(lo, hi, 0,1,2,3,4,5,6,7,8,9,10,11,12,13,14,15);
}

static __device__ __forceinline__ v8f wmma_f16(v16h a, v16h b, v8f c) {
  return __builtin_amdgcn_wmma_f32_16x16x32_f16(false, a, false, b, (short)0, c,
                                                false, false);
}

// max-reduce across a 16-lane half: xor1/2/4 via DPP8 (VALU), xor8 via shuffle
static __device__ __forceinline__ float rowmax16(float x) {
  float y;
  y = __int_as_float(__builtin_amdgcn_mov_dpp8(__float_as_int(x),
        DPP8_SEL(1,0,3,2,5,4,7,6)));
  x = fmaxf(x, y);
  y = __int_as_float(__builtin_amdgcn_mov_dpp8(__float_as_int(x),
        DPP8_SEL(2,3,0,1,6,7,4,5)));
  x = fmaxf(x, y);
  y = __int_as_float(__builtin_amdgcn_mov_dpp8(__float_as_int(x),
        DPP8_SEL(4,5,6,7,0,1,2,3)));
  x = fmaxf(x, y);
  x = fmaxf(x, __shfl_xor(x, 8, 32));   // stays within the 16-lane half
  return x;
}

__global__ __launch_bounds__(128, 1)
void TiledAttention_81166291959865_kernel(const float* __restrict__ qg,
                                          const float* __restrict__ kg,
                                          const float* __restrict__ vg,
                                          float* __restrict__ og) {
  __shared__ _Float16 Ksh[BN][KPAD];           // [key][head], f16
  __shared__ _Float16 Vsh[HD][VPAD];           // [head][key], f16 (transposed V)
  __shared__ _Float16 Psh[NWAVE][16][VPAD];    // per-wave P transpose buffer

  const int tid   = threadIdx.x;
  const int wave  = tid >> 5;
  const int lane  = tid & 31;
  const int laneN = lane & 15;
  const int hf    = lane >> 4;
  const int b0    = hf ? 16 : 0;   // A-layout K-split base for this half

  const int blk  = blockIdx.x;
  const int b    = blk / (NQ / BM);
  const int mblk = blk % (NQ / BM);
  const int m0   = mblk * BM + wave * 16;

  const float* qrow  = qg + ((size_t)b * NQ + (size_t)(m0 + laneN)) * HD;
  const float* kbase = kg + (size_t)b * NQ * HD;
  const float* vbase = vg + (size_t)b * NQ * HD;

  // ---- Q tile (f32 -> f16) in WMMA A layout, 4 chunks of K=32 ----
  v16h Aq[4];
#pragma unroll
  for (int c = 0; c < 4; ++c) {
    f32x4 x0 = *(const f32x4*)(qrow + 32 * c + b0 + 0);
    f32x4 x1 = *(const f32x4*)(qrow + 32 * c + b0 + 4);
    f32x4 x2 = *(const f32x4*)(qrow + 32 * c + b0 + 8);
    f32x4 x3 = *(const f32x4*)(qrow + 32 * c + b0 + 12);
    v16h a;
#pragma unroll
    for (int i = 0; i < 4; ++i) {
      a[i]      = (_Float16)x0[i];
      a[4 + i]  = (_Float16)x1[i];
      a[8 + i]  = (_Float16)x2[i];
      a[12 + i] = (_Float16)x3[i];
    }
    Aq[c] = a;
  }

  v16h onesB;
#pragma unroll
  for (int i = 0; i < 16; ++i) onesB[i] = (_Float16)1.0f;

  v8f  O[8] = {};
  float mt[8], l[8];
#pragma unroll
  for (int j = 0; j < 8; ++j) { mt[j] = -1.0e30f; l[j] = 0.0f; }

  const float sc = 0.088388347648318447f * 1.4426950408889634f; // 1/sqrt(H)*log2e

#pragma unroll 1
  for (int kb = 0; kb < NQ / BN; ++kb) {
    const int key0 = kb * BN;
    __syncthreads();

    // ---- stage K -> Ksh[key][head], V -> Vsh[head][key] (key-pair packed) ----
#pragma unroll
    for (int u = 0; u < 8; ++u) {
      const int p  = tid + 128 * u;      // 1024 pair-units: 32 key-pairs x 32 h4
      const int kp = p >> 5;             // key pair (uniform per wave-instr)
      const int h4 = p & 31;             // f32x4 index along head (lane = h4)
      const size_t g0 = (size_t)(key0 + 2 * kp) * HD + (size_t)h4 * 4;
      f32x4 ka = *(const f32x4*)(kbase + g0);
      f32x4 kb2 = *(const f32x4*)(kbase + g0 + HD);
      f32x4 va = *(const f32x4*)(vbase + g0);
      f32x4 vb2 = *(const f32x4*)(vbase + g0 + HD);
      v4h kha = { (_Float16)ka[0], (_Float16)ka[1], (_Float16)ka[2], (_Float16)ka[3] };
      v4h khb = { (_Float16)kb2[0], (_Float16)kb2[1], (_Float16)kb2[2], (_Float16)kb2[3] };
      *(v4h*)&Ksh[2 * kp][h4 * 4]     = kha;
      *(v4h*)&Ksh[2 * kp + 1][h4 * 4] = khb;
#pragma unroll
      for (int i = 0; i < 4; ++i) {
        v2h t = { (_Float16)va[i], (_Float16)vb2[i] };
        *(v2h*)&Vsh[h4 * 4 + i][2 * kp] = t;
      }
    }
    if (kb + 1 < NQ / BN) {
      const size_t poff = (size_t)(key0 + BN) * HD + (size_t)tid * 4;
      __builtin_prefetch(kbase + poff, 0, 1);
      __builtin_prefetch(vbase + poff, 0, 1);
    }
    __syncthreads();

    // ---- S = Q K^T : 4 tiles (keys 16t..16t+15), K-dim 128 in 4 chunks ----
    v8f S[4] = {};
    v16h Bc[4], Bn[4];
#pragma unroll
    for (int t = 0; t < 4; ++t) {
      const _Float16* r = &Ksh[16 * t + laneN][hf * 16];
      Bc[t] = cat8(*(const v8h*)r, *(const v8h*)(r + 8));
    }
#pragma unroll
    for (int hc = 0; hc < 4; ++hc) {
      if (hc < 3) {
        const int ho = (hc + 1) * 32 + hf * 16;
#pragma unroll
        for (int t = 0; t < 4; ++t) {
          const _Float16* r = &Ksh[16 * t + laneN][ho];
          Bn[t] = cat8(*(const v8h*)r, *(const v8h*)(r + 8));
        }
      }
#pragma unroll
      for (int t = 0; t < 4; ++t)
        S[t] = wmma_f16(Aq[hc], Bc[t], S[t]);
#pragma unroll
      for (int t = 0; t < 4; ++t) Bc[t] = Bn[t];
    }

    // ---- online softmax; P overwrites S ----
    float alpha[8];
#pragma unroll
    for (int j = 0; j < 8; ++j) {
      float cur = fmaxf(fmaxf(S[0][j], S[1][j]), fmaxf(S[2][j], S[3][j]));
      cur = rowmax16(cur);
      const float mn = fmaxf(mt[j], cur * sc);
      alpha[j] = __builtin_amdgcn_exp2f(mt[j] - mn);
      mt[j] = mn;
#pragma unroll
      for (int t = 0; t < 4; ++t)
        S[t][j] = __builtin_amdgcn_exp2f(S[t][j] * sc - mn);
    }
#pragma unroll
    for (int t = 0; t < 8; ++t)
#pragma unroll
      for (int j = 0; j < 8; ++j)
        O[t][j] *= alpha[j];

    // ---- transpose P (C layout -> A layout) via per-wave LDS buffer ----
    _Float16* Pw = &Psh[wave][0][0];
#pragma unroll
    for (int j = 0; j < 8; ++j) {
      const int r = (j + 8 * hf) * VPAD + laneN;
#pragma unroll
      for (int t = 0; t < 4; ++t)
        Pw[r + 16 * t] = (_Float16)S[t][j];
    }
    asm volatile("s_wait_dscnt 0" ::: "memory");   // wave-private buffer
    v16h Ap[2];
#pragma unroll
    for (int c = 0; c < 2; ++c) {
      const _Float16* r = &Pw[laneN * VPAD + 32 * c + b0];
      Ap[c] = cat8(*(const v8h*)r, *(const v8h*)(r + 8));
    }

    // ---- row sums of P via WMMA against ones (replaces lane shuffles) ----
    v8f R = {};
    R = wmma_f16(Ap[0], onesB, R);
    R = wmma_f16(Ap[1], onesB, R);
#pragma unroll
    for (int j = 0; j < 8; ++j) l[j] = l[j] * alpha[j] + R[j];

    // ---- O += P * V : 8 output tiles x 2 key-chunks ----
    v16h Vc[2], Vn[2];
#pragma unroll
    for (int c = 0; c < 2; ++c) {
      const _Float16* r = &Vsh[laneN][32 * c + hf * 16];
      Vc[c] = cat8(*(const v8h*)r, *(const v8h*)(r + 8));
    }
#pragma unroll
    for (int t = 0; t < 8; ++t) {
      if (t < 7) {
#pragma unroll
        for (int c = 0; c < 2; ++c) {
          const _Float16* r = &Vsh[16 * (t + 1) + laneN][32 * c + hf * 16];
          Vn[c] = cat8(*(const v8h*)r, *(const v8h*)(r + 8));
        }
      }
      O[t] = wmma_f16(Ap[0], Vc[0], O[t]);
      O[t] = wmma_f16(Ap[1], Vc[1], O[t]);
      Vc[0] = Vn[0]; Vc[1] = Vn[1];
    }
  }

  // ---- epilogue: out = O / l ----
  float* orow = og + ((size_t)b * NQ + (size_t)m0) * HD;
#pragma unroll
  for (int j = 0; j < 8; ++j) {
    const float inv = 1.0f / l[j];
    float* op = orow + (size_t)(j + 8 * hf) * HD + laneN;
#pragma unroll
    for (int t = 0; t < 8; ++t)
      op[t * 16] = O[t][j] * inv;
  }
}

extern "C" void kernel_launch(void* const* d_in, const int* in_sizes, int n_in,
                              void* d_out, int out_size, void* d_ws, size_t ws_size,
                              hipStream_t stream) {
  (void)in_sizes; (void)n_in; (void)out_size; (void)d_ws; (void)ws_size;
  const float* q = (const float*)d_in[0];
  const float* k = (const float*)d_in[1];
  const float* v = (const float*)d_in[2];
  float* out = (float*)d_out;
  dim3 grid(NB * (NQ / BM));   // 256 workgroups
  dim3 block(128);             // 4 wave32 per workgroup
  hipLaunchKernelGGL(TiledAttention_81166291959865_kernel, grid, block, 0, stream,
                     q, k, v, out);
}
